// NER_14087492731546
// MI455X (gfx1250) — matrix operand order repaired and checked by reading, not compiled, not run on hardware
//
#include <hip/hip_runtime.h>

typedef __attribute__((ext_vector_type(16))) __bf16 v16bf;
typedef __attribute__((ext_vector_type(8)))  __bf16 v8bf;
typedef __attribute__((ext_vector_type(4)))  __bf16 v4bf;
typedef __attribute__((ext_vector_type(2)))  __bf16 v2bf;
typedef __attribute__((ext_vector_type(8)))  float  v8f;

// GCC-style vector to match the async-LDS builtin's parameter type exactly:
//   '__attribute__((__vector_size__(4 * sizeof(int)))) int __device__ *'
typedef int v4i_g __attribute__((vector_size(16)));
typedef __attribute__((address_space(1))) v4i_g* g1_v4i;  // global
typedef __attribute__((address_space(3))) v4i_g* l3_v4i;  // LDS

#define B_   32
#define S_   512
#define H_   1024
#define T_   37
#define LP_  39
#define NTOK (B_ * S_)

#if __has_builtin(__builtin_amdgcn_global_load_async_to_lds_b128) && \
    __has_builtin(__builtin_amdgcn_s_wait_asynccnt)
#define HAVE_ASYNC_LDS 1
#endif

__device__ __forceinline__ void bf16_split(float x, __bf16& hi, __bf16& lo) {
  hi = (__bf16)x;
  lo = (__bf16)(x - (float)hi);
}

// ---------------------------------------------------------------------------
// Kernel 1: scores = (h @ bio^T)/sqrt(H) -> softmax(T) -> aware = attn @ bio
// bio_embed (151.5 KB) + 16 h rows (64 KB) live in LDS (CDNA5: 320 KB/WG).
// Staging uses GLOBAL_LOAD_ASYNC_TO_LDS_B128 (ASYNCcnt) when available.
// ---------------------------------------------------------------------------
#define TPB1 256
#define TOKB 16
__global__ void k_aware(const float* __restrict__ h,
                        const float* __restrict__ bio,
                        float* __restrict__ aware) {
  extern __shared__ float smem[];
  float* sBio = smem;                    // T_*H_
  float* sH   = sBio + T_ * H_;          // TOKB*H_
  float* sSc  = sH + TOKB * H_;          // TOKB*40
  const int tid  = threadIdx.x;
  const int wv   = tid >> 5;
  const int lane = tid & 31;
  const long tok0 = (long)blockIdx.x * TOKB;

#ifdef HAVE_ASYNC_LDS
  for (int i = tid; i < (T_ * H_) / 4; i += TPB1)
    __builtin_amdgcn_global_load_async_to_lds_b128(
        (g1_v4i)(bio + 4 * i), (l3_v4i)(sBio + 4 * i), 0, 0);
  for (int i = tid; i < (TOKB * H_) / 4; i += TPB1)
    __builtin_amdgcn_global_load_async_to_lds_b128(
        (g1_v4i)(h + tok0 * H_ + 4 * i), (l3_v4i)(sH + 4 * i), 0, 0);
  __builtin_amdgcn_s_wait_asynccnt(0);
#else
  for (int i = tid; i < T_ * H_; i += TPB1) sBio[i] = bio[i];
  for (int i = tid; i < TOKB * H_; i += TPB1) sH[i] = h[tok0 * H_ + i];
#endif
  __syncthreads();

  // scores: one wave per (token, t) dot product of length H
  const float scale = 0.03125f;  // 1/sqrt(1024)
  for (int idx = wv; idx < TOKB * T_; idx += 8) {
    int tok = idx / T_;
    int t   = idx - tok * T_;
    float p = 0.f;
    for (int k = lane; k < H_; k += 32) p += sH[tok * H_ + k] * sBio[t * H_ + k];
    for (int off = 16; off > 0; off >>= 1) p += __shfl_xor(p, off, 32);
    if (lane == 0) sSc[tok * 40 + t] = p * scale;
  }
  __syncthreads();

  // softmax over T (37 > 32 lanes -> each lane holds t and t+32)
  for (int tok = wv; tok < TOKB; tok += 8) {
    float v0 = (lane < T_)      ? sSc[tok * 40 + lane]      : -3.4e38f;
    float v1 = (lane + 32 < T_) ? sSc[tok * 40 + lane + 32] : -3.4e38f;
    float m = fmaxf(v0, v1);
    for (int off = 16; off > 0; off >>= 1) m = fmaxf(m, __shfl_xor(m, off, 32));
    float e0 = (lane < T_)      ? expf(v0 - m) : 0.f;
    float e1 = (lane + 32 < T_) ? expf(v1 - m) : 0.f;
    float s = e0 + e1;
    for (int off = 16; off > 0; off >>= 1) s += __shfl_xor(s, off, 32);
    float inv = 1.f / s;
    if (lane < T_)      sSc[tok * 40 + lane]      = e0 * inv;
    if (lane + 32 < T_) sSc[tok * 40 + lane + 32] = e1 * inv;
  }
  __syncthreads();

  // aware[tok, j] = sum_t attn[t] * bio[t, j]
  for (int tok = 0; tok < TOKB; tok++) {
    for (int jj = 0; jj < H_ / TPB1; jj++) {
      int j = tid + jj * TPB1;
      float acc = 0.f;
      for (int t = 0; t < T_; t++) acc += sSc[tok * 40 + t] * sBio[t * H_ + j];
      aware[(tok0 + tok) * H_ + j] = acc;
    }
  }
}

// ---------------------------------------------------------------------------
// Kernel 2: x2 = gelu([h|aware|h*aware] @ W_cat + b_cat)  (M=16384,N=1024,K=3072)
// bf16x3 fp32-emulation on v_wmma_f32_16x16x32_bf16. hi/lo split is done once
// at staging time; tiles stored as packed bf16 (B transposed to [n][k]) so each
// fragment is two ds_load_b128's. Row stride 40 bf16 (80B, 16B-aligned) makes
// the 16-lane b128 fragment reads tile all 64 LDS banks conflict-free.
// ---------------------------------------------------------------------------
#define BM 128
#define BN 64
#define BK 32
#define LDK 40

__global__ __launch_bounds__(256) void k_gemm_cat(
    const float* __restrict__ h, const float* __restrict__ aware,
    const float* __restrict__ Wcat, const float* __restrict__ bcat,
    float* __restrict__ x2) {
  __shared__ __bf16 sAh[BM * LDK];
  __shared__ __bf16 sAl[BM * LDK];
  __shared__ __bf16 sBh[BN * LDK];
  __shared__ __bf16 sBl[BN * LDK];
  const int tid  = threadIdx.x;
  const int wv   = tid >> 5;
  const int lane = tid & 31;
  const int r    = lane & 15;   // row (A) / col (B,C)
  const int hf   = lane >> 4;   // lane-half selects K/M sub-blocks
  const long m0 = (long)blockIdx.y * BM;
  const int  n0 = blockIdx.x * BN;

  v8f acc[4];
#pragma unroll
  for (int nt = 0; nt < 4; nt++)
#pragma unroll
    for (int j = 0; j < 8; j++) acc[nt][j] = 0.f;

  for (int k0 = 0; k0 < 3 * H_; k0 += BK) {
    const int seg = k0 >> 10;        // 0: h, 1: aware, 2: h*aware
    const int kk0 = k0 & (H_ - 1);

    // stage A tile (on-the-fly concat feature), split to bf16 hi/lo
    for (int i = tid; i < (BM * BK) / 4; i += 256) {  // 1024 float4's
      int row = i >> 3;
      int c4  = (i & 7) << 2;
      long base = (m0 + row) * H_ + kk0 + c4;
      float4 v;
      if (seg == 0)      v = *(const float4*)(h + base);
      else if (seg == 1) v = *(const float4*)(aware + base);
      else {
        float4 a = *(const float4*)(h + base);
        float4 w = *(const float4*)(aware + base);
        v = make_float4(a.x * w.x, a.y * w.y, a.z * w.z, a.w * w.w);
      }
      float vv[4] = {v.x, v.y, v.z, v.w};
      v4bf hi, lo;
#pragma unroll
      for (int c = 0; c < 4; c++) {
        __bf16 hh, ll; bf16_split(vv[c], hh, ll);
        hi[c] = hh; lo[c] = ll;
      }
      *(v4bf*)&sAh[row * LDK + c4] = hi;
      *(v4bf*)&sAl[row * LDK + c4] = lo;
    }
    // stage B tile transposed: sB[n][k] = W_cat[k0+k][n0+n], bf16 hi/lo
    for (int i = tid; i < (BK * BN) / 2; i += 256) {  // 1024 k-pairs
      int n  = i & 63;
      int kp = i >> 6;  // 0..15 -> k = 2*kp, 2*kp+1
      long g = (long)(k0 + 2 * kp) * H_ + n0 + n;
      float w0 = Wcat[g];
      float w1 = Wcat[g + H_];
      __bf16 h0, l0, h1, l1;
      bf16_split(w0, h0, l0);
      bf16_split(w1, h1, l1);
      v2bf hi, lo;
      hi[0] = h0; hi[1] = h1;
      lo[0] = l0; lo[1] = l1;
      *(v2bf*)&sBh[n * LDK + 2 * kp] = hi;
      *(v2bf*)&sBl[n * LDK + 2 * kp] = lo;
    }
    __syncthreads();

    // A fragment: elems 0-7 = K hf*8.., elems 8-15 = K 16+hf*8.. (ISA 7.12.2)
    const int arow = (wv * 16 + r) * LDK;
    v8bf a0h = *(const v8bf*)&sAh[arow + hf * 8];
    v8bf a1h = *(const v8bf*)&sAh[arow + 16 + hf * 8];
    v8bf a0l = *(const v8bf*)&sAl[arow + hf * 8];
    v8bf a1l = *(const v8bf*)&sAl[arow + 16 + hf * 8];
    v16bf ahi = __builtin_shufflevector(a0h, a1h, 0,1,2,3,4,5,6,7,8,9,10,11,12,13,14,15);
    v16bf alo = __builtin_shufflevector(a0l, a1l, 0,1,2,3,4,5,6,7,8,9,10,11,12,13,14,15);
#pragma unroll
    for (int nt = 0; nt < 4; nt++) {
      const int brow = (nt * 16 + r) * LDK + hf * 8;  // elems 0-7 = K hf*8.., 8-15 = K 16+hf*8..
      v8bf b0h = *(const v8bf*)&sBh[brow];
      v8bf b1h = *(const v8bf*)&sBh[brow + 16];
      v8bf b0l = *(const v8bf*)&sBl[brow];
      v8bf b1l = *(const v8bf*)&sBl[brow + 16];
      v16bf bhi = __builtin_shufflevector(b0h, b1h, 0,1,2,3,4,5,6,7,8,9,10,11,12,13,14,15);
      v16bf blo = __builtin_shufflevector(b0l, b1l, 0,1,2,3,4,5,6,7,8,9,10,11,12,13,14,15);
      acc[nt] = __builtin_amdgcn_wmma_f32_16x16x32_bf16(
          false, ahi, false, bhi, (short)0, acc[nt], false, false);
      acc[nt] = __builtin_amdgcn_wmma_f32_16x16x32_bf16(
          false, ahi, false, blo, (short)0, acc[nt], false, false);
      acc[nt] = __builtin_amdgcn_wmma_f32_16x16x32_bf16(
          false, alo, false, bhi, (short)0, acc[nt], false, false);
    }
    __syncthreads();
  }

  // epilogue: + b_cat, exact GELU, store x2
#pragma unroll
  for (int nt = 0; nt < 4; nt++) {
#pragma unroll
    for (int j = 0; j < 8; j++) {
      long m = m0 + wv * 16 + hf * 8 + j;   // D layout: VGPR j = row hf*8+j
      int  n = n0 + nt * 16 + r;
      float v = acc[nt][j] + bcat[n];
      float g = 0.5f * v * (1.f + erff(v * 0.70710678118f));
      x2[m * H_ + n] = g;
    }
  }
}

// ---------------------------------------------------------------------------
// Kernel 3: ner_scores[:, :37] = x2 @ W_crf + b_crf ; [:, 37:39] = -10000
// One wave per token (N=37 too skinny for WMMA; bound by the 64 MB x2 read).
// ---------------------------------------------------------------------------
__global__ void k_logits(const float* __restrict__ x2,
                         const float* __restrict__ Wcrf,
                         const float* __restrict__ bcrf,
                         float* __restrict__ outScores) {
  const int wv   = threadIdx.x >> 5;
  const int lane = threadIdx.x & 31;
  const long tok = (long)blockIdx.x * 8 + wv;
  float xr[32];
#pragma unroll
  for (int i = 0; i < 32; i++) xr[i] = x2[tok * H_ + lane + 32 * i];
  for (int t = 0; t < T_; t++) {
    float p = 0.f;
#pragma unroll
    for (int i = 0; i < 32; i++) p += xr[i] * Wcrf[(lane + 32 * i) * T_ + t];
    for (int off = 16; off > 0; off >>= 1) p += __shfl_xor(p, off, 32);
    if (lane == 0) outScores[tok * LP_ + t] = p + bcrf[t];
  }
  if (lane < 2) outScores[tok * LP_ + T_ + lane] = -10000.0f;
}

// ---------------------------------------------------------------------------
// Kernel 4: CRF log-likelihood. One block (64 thr) per batch row; transition
// matrix in LDS; 39-state forward scan clipped at len (alpha frozen beyond).
// ---------------------------------------------------------------------------
__global__ void k_crf(const float* __restrict__ scores,
                      const int* __restrict__ labels,
                      const int* __restrict__ lens,
                      const float* __restrict__ trans,
                      float* __restrict__ loss) {
  __shared__ float sT[LP_ * LP_];
  __shared__ float sAlpha[LP_];
  __shared__ float sNew[LP_];
  __shared__ float sRed[64];
  const int tid = threadIdx.x;
  const int b   = blockIdx.x;
  const int len = lens[b];

  for (int i = tid; i < LP_ * LP_; i += 64) sT[i] = trans[i];

  // gold: unary part (doesn't touch sT)
  float g = 0.f;
  for (int t = tid; t < S_; t += 64)
    if (t < len) g += scores[((long)b * S_ + t) * LP_ + labels[b * S_ + t]];
  __syncthreads();  // sT ready
  // gold: transition part over p = 0..len  (ext[p]->ext[p+1])
  for (int p = tid; p <= len; p += 64) {
    int frm = (p == 0) ? T_ : labels[b * S_ + p - 1];
    int to  = (p + 1 <= len) ? labels[b * S_ + p] : (T_ + 1);
    g += sT[to * LP_ + frm];
  }
  sRed[tid] = g;
  __syncthreads();
  for (int off = 32; off > 0; off >>= 1) {
    if (tid < off) sRed[tid] += sRed[tid + off];
    __syncthreads();
  }
  float gold = sRed[0];

  if (tid < LP_) sAlpha[tid] = (tid == T_) ? 0.f : -100.f;
  __syncthreads();
  for (int t = 0; t < len; t++) {
    if (tid < LP_) {
      const float* row = &sT[tid * LP_];
      float m = -3.4e38f;
      for (int j = 0; j < LP_; j++) m = fmaxf(m, sAlpha[j] + row[j]);
      float s = 0.f;
      for (int j = 0; j < LP_; j++) s += expf(sAlpha[j] + row[j] - m);
      sNew[tid] = scores[((long)b * S_ + t) * LP_ + tid] + m + logf(s);
    }
    __syncthreads();
    if (tid < LP_) sAlpha[tid] = sNew[tid];
    __syncthreads();
  }
  if (tid == 0) {
    float m = -3.4e38f;
    for (int i = 0; i < LP_; i++)
      m = fmaxf(m, sAlpha[i] + sT[(T_ + 1) * LP_ + i]);
    float s = 0.f;
    for (int i = 0; i < LP_; i++)
      s += expf(sAlpha[i] + sT[(T_ + 1) * LP_ + i] - m);
    loss[b] = gold - (m + logf(s));
  }
}

// ---------------------------------------------------------------------------
extern "C" void kernel_launch(void* const* d_in, const int* in_sizes, int n_in,
                              void* d_out, int out_size, void* d_ws, size_t ws_size,
                              hipStream_t stream) {
  (void)in_sizes; (void)n_in; (void)out_size; (void)ws_size;
  const float* h      = (const float*)d_in[0];
  // d_in[1] = token_masks (unused by the reference computation)
  const int*   lens   = (const int*)  d_in[2];
  const int*   labels = (const int*)  d_in[3];
  const float* bio    = (const float*)d_in[4];
  const float* Wcat   = (const float*)d_in[5];
  const float* bcat   = (const float*)d_in[6];
  const float* Wcrf   = (const float*)d_in[7];
  const float* bcrf   = (const float*)d_in[8];
  const float* trans  = (const float*)d_in[9];

  float* outScores = (float*)d_out;                 // NTOK * LP_
  float* outLoss   = outScores + (long)NTOK * LP_;  // B_

  // workspace: aware (64 MB) + x2 (64 MB)
  float* aware = (float*)d_ws;
  float* x2    = aware + (long)NTOK * H_;

  size_t smem1 = (size_t)(T_ * H_ + TOKB * H_ + TOKB * 40) * sizeof(float);
  hipLaunchKernelGGL(k_aware, dim3(NTOK / TOKB), dim3(TPB1), smem1, stream,
                     h, bio, aware);
  hipLaunchKernelGGL(k_gemm_cat, dim3(H_ / BN, NTOK / BM), dim3(256), 0, stream,
                     h, aware, Wcat, bcat, x2);
  hipLaunchKernelGGL(k_logits, dim3(NTOK / 8), dim3(256), 0, stream,
                     x2, Wcrf, bcrf, outScores);
  hipLaunchKernelGGL(k_crf, dim3(B_), dim3(64), 0, stream,
                     outScores, labels, lens, trans, outLoss);
}
